// MotionCompensator_37177236914860
// MI455X (gfx1250) — compile-verified, matrix operand order checked by reading, not run on hardware
//
#include <hip/hip_runtime.h>

typedef __attribute__((ext_vector_type(16))) _Float16 v16h;
typedef __attribute__((ext_vector_type(8)))  float    v8f;

// Problem dims are fixed by the reference — make them constexpr so channel
// strides fold into 24-bit instruction immediates.
constexpr int BN = 2, H = 192, W = 192;
constexpr int HW = H * W;

// ---------------------------------------------------------------------------
// WMMA 16-bit A/B fragment K mapping (ISA 7.12.2): lane L holds row/col L&15,
// kbase=(L>>4)*8, element j <-> K = j<8 ? kb+j : 16+kb+(j-8).
// GEMM K dimension is TAP-MAJOR:  k' = kk*Cin + c   (kk in [0,9), c in [0,Cin)).
// Cin is a power of two, so a 32-wide K chunk has ONE tap and 32 consecutive
// channels, and per lane the 16 elements are channels {cb..cb+7, cb+16..cb+23}.
// ---------------------------------------------------------------------------
__device__ __host__ inline int kmap(int kb, int j) {
    return (j < 8) ? (kb + j) : (16 + kb + (j - 8));
}

// Pack fp32 weights (Cout, Cin, 3, 3) into f16 B-fragments in tap-major order:
// Wp[((kt*Ntiles + nt)*32 + lane)*16 + j] = W[n][c][kk]   with
//   k' = kt*32 + kmap(kb,j),  c = k' % Cin,  kk = k' / Cin,  n = nt*16 + lane&15
__global__ void pack_w_kernel(const float* __restrict__ Wsrc,
                              _Float16* __restrict__ Wp,
                              int Ntiles, int Ktiles, int Cin) {
    int idx = blockIdx.x * blockDim.x + threadIdx.x;
    int total = Ktiles * Ntiles * 32 * 16;
    if (idx >= total) return;
    int j    = idx & 15;
    int lane = (idx >> 4) & 31;
    int rest = idx >> 9;
    int nt   = rest % Ntiles;
    int kt   = rest / Ntiles;
    int kb   = (lane >> 4) * 8;
    int kp   = kt * 32 + kmap(kb, j);
    int c    = kp % Cin;
    int kk   = kp / Cin;
    int n    = nt * 16 + (lane & 15);
    Wp[idx] = (_Float16)Wsrc[((size_t)n * Cin + c) * 9 + kk];
}

// ---------------------------------------------------------------------------
// 3x3 pad-1 conv as implicit GEMM via WMMA. One wave per block; M-tile = 16
// consecutive w positions of one (b,h) row. Input = channel-concat X1|X2.
// ---------------------------------------------------------------------------
template<int NT, int C1, int C2>
__global__ void conv3x3_wmma_kernel(const float* __restrict__ X1,
                                    const float* __restrict__ X2,
                                    const _Float16* __restrict__ Wp,
                                    const float* __restrict__ bias,
                                    const float* __restrict__ residual,
                                    float* __restrict__ Out, int act) {
    constexpr int Cin  = C1 + C2;
    constexpr int KT   = (Cin * 9) / 32;   // 18 or 36
    constexpr int Cout = NT * 16;
    constexpr int wt   = W / 16;

    int tile = blockIdx.x;
    int w0   = (tile % wt) * 16;
    int h    = (tile / wt) % H;
    int b    =  tile / (wt * H);

    int lane = threadIdx.x & 31;
    int kb   = (lane >> 4) * 8;
    int wpos = w0 + (lane & 15);

    v8f zero = {0.f,0.f,0.f,0.f,0.f,0.f,0.f,0.f};
    v8f acc[NT];
#pragma unroll
    for (int nt = 0; nt < NT; ++nt) acc[nt] = zero;

    const _Float16* wb = Wp;
    for (int kt = 0; kt < KT; ++kt) {
        int kk    = (kt * 32) / Cin;       // power-of-two -> shift
        int cbase = (kt * 32) & (Cin - 1);
        int ky = kk / 3;
        int kx = kk - ky * 3;
        int iy = h + ky - 1;               // one tap per chunk
        int ix = wpos + kx - 1;
        bool valid = (iy >= 0) && (iy < H) && (ix >= 0) && (ix < W);

        // Chunk's 32 channels live entirely in one of the concat tensors
        // (C1 is a multiple of 32), so selection is uniform per chunk.
        const float* src;
        int cb, Csel;
        if (cbase < C1) { src = X1; cb = cbase;      Csel = C1; }
        else            { src = X2; cb = cbase - C1; Csel = C2 ? C2 : 1; }
        const float* p = src + ((size_t)(b * Csel + cb + kb) * H + iy) * W + ix;

        v16h a;
#pragma unroll
        for (int j = 0; j < 16; ++j) a[j] = (_Float16)0.f;
        if (valid) {
#pragma unroll
            for (int j = 0; j < 16; ++j) {
                int coff = (j < 8) ? j : (j + 8);      // kmap - kb
                a[j] = (_Float16)p[(size_t)coff * HW]; // immediate offsets
            }
        }

#pragma unroll
        for (int nt = 0; nt < NT; ++nt) {
            v16h bf = *(const v16h*)(wb + ((size_t)nt * 32 + lane) * 16);
            acc[nt] = __builtin_amdgcn_wmma_f32_16x16x32_f16(
                false, a, false, bf, (short)0, acc[nt], false, false);
        }
        wb += NT * 512;
    }

    // C/D layout: lane L -> N = L&15, row M = vgpr + 8*(L>>4).
    int n0    = lane & 15;
    int mbase = (lane >> 4) * 8;
#pragma unroll
    for (int nt = 0; nt < NT; ++nt) {
        int n = nt * 16 + n0;
        float bi = bias[n];
#pragma unroll
        for (int v = 0; v < 8; ++v) {
            size_t oi = (((size_t)b * Cout + n) * H + h) * (size_t)W
                        + (w0 + mbase + v);
            float val = acc[nt][v] + bi;
            if (act) val = (val > 0.f) ? val : 0.1f * val;
            if (residual) val += residual[oi];
            Out[oi] = val;
        }
    }
}

// ---------------------------------------------------------------------------
// Deformable conv (groups=8, Cg=8, Cin=64, Cout=64, K=576 tap-major).
// Per K-chunk each lane touches exactly TWO aligned 8-channel groups, each
// with one deform group g -> 2 bilinear coordinate setups per chunk, then
// 8 mask-weighted 4-corner gathers with immediate channel offsets.
// Off layout (B,144,H,W), channel = (g*9+kk)*2 + {0:dy, 1:dx}.
// ---------------------------------------------------------------------------
__global__ void deform_wmma_kernel(const float* __restrict__ X,
                                   const float* __restrict__ Off,
                                   const _Float16* __restrict__ Wp,
                                   const float* __restrict__ bias,
                                   float* __restrict__ Out) {
    constexpr int NT = 4;     // Cout = 64
    constexpr int KT = 18;    // K = 576
    constexpr int wt = W / 16;

    int tile = blockIdx.x;
    int w0   = (tile % wt) * 16;
    int h    = (tile / wt) % H;
    int b    =  tile / (wt * H);

    int lane = threadIdx.x & 31;
    int kb   = (lane >> 4) * 8;
    int wpos = w0 + (lane & 15);

    const float* Xb   = X   + (size_t)b * 64  * HW;
    const float* Offb = Off + (size_t)b * 144 * HW + (size_t)h * W + wpos;

    v8f zero = {0.f,0.f,0.f,0.f,0.f,0.f,0.f,0.f};
    v8f acc[NT];
#pragma unroll
    for (int nt = 0; nt < NT; ++nt) acc[nt] = zero;

    const _Float16* wb = Wp;
    for (int kt = 0; kt < KT; ++kt) {
        int kk    = kt >> 1;          // (kt*32)/64
        int cbase = (kt & 1) * 32;
        int ky = kk / 3;
        int kx = kk - ky * 3;

        v16h a;
#pragma unroll
        for (int half = 0; half < 2; ++half) {
            int c0 = cbase + kb + half * 16;   // aligned group of 8 channels
            int g  = c0 >> 3;
            int chdy = (g * 9 + kk) * 2;
            float dy = Offb[(size_t)chdy * HW];
            float dx = Offb[(size_t)(chdy + 1) * HW];
            float py = (float)(h - 1 + ky)    + dy;
            float px = (float)(wpos - 1 + kx) + dx;
            float fy = floorf(py), fx = floorf(px);
            float wy1 = py - fy, wx1 = px - fx;
            float wy0 = 1.f - wy1, wx0 = 1.f - wx1;
            int y0 = (int)fy, x0 = (int)fx;
            int y1 = y0 + 1,  x1 = x0 + 1;
            bool vy0 = (y0 >= 0) && (y0 < H);
            bool vy1 = (y1 >= 0) && (y1 < H);
            bool vx0 = (x0 >= 0) && (x0 < W);
            bool vx1 = (x1 >= 0) && (x1 < W);
            int cy0 = min(max(y0, 0), H - 1), cy1 = min(max(y1, 0), H - 1);
            int cx0 = min(max(x0, 0), W - 1), cx1 = min(max(x1, 0), W - 1);
            float w00 = (vy0 && vx0) ? wy0 * wx0 : 0.f;
            float w01 = (vy0 && vx1) ? wy0 * wx1 : 0.f;
            float w10 = (vy1 && vx0) ? wy1 * wx0 : 0.f;
            float w11 = (vy1 && vx1) ? wy1 * wx1 : 0.f;
            const float* pc  = Xb + (size_t)c0 * HW;
            const float* p00 = pc + (size_t)cy0 * W + cx0;
            const float* p01 = pc + (size_t)cy0 * W + cx1;
            const float* p10 = pc + (size_t)cy1 * W + cx0;
            const float* p11 = pc + (size_t)cy1 * W + cx1;
#pragma unroll
            for (int t = 0; t < 8; ++t) {
                float s = p00[(size_t)t * HW] * w00 + p01[(size_t)t * HW] * w01
                        + p10[(size_t)t * HW] * w10 + p11[(size_t)t * HW] * w11;
                a[half * 8 + t] = (_Float16)s;
            }
        }

#pragma unroll
        for (int nt = 0; nt < NT; ++nt) {
            v16h bf = *(const v16h*)(wb + ((size_t)nt * 32 + lane) * 16);
            acc[nt] = __builtin_amdgcn_wmma_f32_16x16x32_f16(
                false, a, false, bf, (short)0, acc[nt], false, false);
        }
        wb += NT * 512;
    }

    int n0    = lane & 15;
    int mbase = (lane >> 4) * 8;
#pragma unroll
    for (int nt = 0; nt < NT; ++nt) {
        int n = nt * 16 + n0;
        float bi = bias[n];
#pragma unroll
        for (int v = 0; v < 8; ++v) {
            size_t oi = (((size_t)b * 64 + n) * H + h) * (size_t)W
                        + (w0 + mbase + v);
            Out[oi] = acc[nt][v] + bi;
        }
    }
}

// ---------------------------------------------------------------------------
extern "C" void kernel_launch(void* const* d_in, const int* in_sizes, int n_in,
                              void* d_out, int out_size, void* d_ws, size_t ws_size,
                              hipStream_t stream) {
    const float* prev    = (const float*)d_in[0]; // (2,64,192,192)
    const float* offsets = (const float*)d_in[1]; // (2,64,192,192)
    const float* w_off   = (const float*)d_in[2]; // (144,64,3,3)
    const float* b_off   = (const float*)d_in[3];
    const float* w_dcn   = (const float*)d_in[4]; // (64,64,3,3)
    const float* b_dcn   = (const float*)d_in[5];
    const float* w_r1    = (const float*)d_in[6]; // (64,128,3,3)
    const float* b_r1    = (const float*)d_in[7];
    const float* w_r2    = (const float*)d_in[8]; // (64,64,3,3)
    const float* b_r2    = (const float*)d_in[9];
    (void)in_sizes; (void)n_in; (void)out_size; (void)ws_size;

    // Carve workspace (256B aligned chunks).
    char* ws = (char*)d_ws;
    size_t p = 0;
    auto carve = [&](size_t bytes) -> char* {
        p = (p + 255) & ~(size_t)255;
        char* r = ws + p;
        p += bytes;
        return r;
    };
    float*    off_buf = (float*)    carve((size_t)BN * 144 * HW * 4); // 42.5 MB
    float*    aligned = (float*)    carve((size_t)BN * 64  * HW * 4); // 18.9 MB
    float*    r1_buf  = (float*)    carve((size_t)BN * 64  * HW * 4); // 18.9 MB
    _Float16* wp_off  = (_Float16*) carve((size_t)18 * 9 * 512 * 2);
    _Float16* wp_dcn  = (_Float16*) carve((size_t)18 * 4 * 512 * 2);
    _Float16* wp_r1   = (_Float16*) carve((size_t)36 * 4 * 512 * 2);
    _Float16* wp_r2   = (_Float16*) carve((size_t)18 * 4 * 512 * 2);

    // 1) Pack weights into tap-major WMMA B-fragment layout (f16).
    {
        int t;
        t = 18 * 9 * 512;
        pack_w_kernel<<<(t + 255) / 256, 256, 0, stream>>>(w_off, wp_off, 9, 18, 64);
        t = 18 * 4 * 512;
        pack_w_kernel<<<(t + 255) / 256, 256, 0, stream>>>(w_dcn, wp_dcn, 4, 18, 64);
        t = 36 * 4 * 512;
        pack_w_kernel<<<(t + 255) / 256, 256, 0, stream>>>(w_r1,  wp_r1,  4, 36, 128);
        t = 18 * 4 * 512;
        pack_w_kernel<<<(t + 255) / 256, 256, 0, stream>>>(w_r2,  wp_r2,  4, 18, 64);
    }

    const int mtiles = BN * H * (W / 16); // 4608 tiles, one wave each

    // 2) Offset conv: 64 -> 144 channels, no activation.
    conv3x3_wmma_kernel<9, 64, 0><<<mtiles, 32, 0, stream>>>(
        offsets, offsets, wp_off, b_off, nullptr, off_buf, 0);

    // 3) Deformable conv: prev + off -> aligned (64 ch).
    deform_wmma_kernel<<<mtiles, 32, 0, stream>>>(
        prev, off_buf, wp_dcn, b_dcn, aligned);

    // 4) r1 = leaky_relu(conv([aligned, prev], w_r1) + b_r1)   (128 -> 64)
    conv3x3_wmma_kernel<4, 64, 64><<<mtiles, 32, 0, stream>>>(
        aligned, prev, wp_r1, b_r1, nullptr, r1_buf, 1);

    // 5) out = aligned + leaky_relu(conv(r1, w_r2) + b_r2)     (64 -> 64)
    conv3x3_wmma_kernel<4, 64, 0><<<mtiles, 32, 0, stream>>>(
        r1_buf, r1_buf, wp_r2, b_r2, aligned, (float*)d_out, 1);
}
